// VisionMamba_46514495816372
// MI455X (gfx1250) — matrix operand order repaired and checked by reading, not compiled, not run on hardware
//
#include <hip/hip_runtime.h>
#include <hip/hip_bf16.h>

// ---------------- model constants ----------------
constexpr int BATCH = 32;
constexpr int C_IN  = 3;
constexpr int IMG   = 224;
constexpr int PTCH  = 16;
constexpr int D     = 384;
constexpr int DEPTH = 24;
constexpr int NCLS  = 1000;
constexpr int DI    = 768;     // 2*D
constexpr int DS    = 16;
constexpr int DCONV = 4;
constexpr int DTR   = 24;
constexpr int GRID  = IMG / PTCH;          // 14
constexpr int NPATCH = GRID * GRID;        // 196
constexpr int L     = NPATCH + 1;          // 197
constexpr int BL    = BATCH * L;           // 6304  (= 197*32)
constexpr int BP    = BATCH * NPATCH;      // 6272  (= 196*32)
constexpr int KIM   = C_IN * PTCH * PTCH;  // 768
constexpr int NXP   = 64;                  // x_proj out padded 56 -> 64
constexpr int KDT   = 32;                  // dt_proj K padded 24 -> 32

typedef __attribute__((ext_vector_type(16))) _Float16 v16h;
typedef __attribute__((ext_vector_type(8)))  _Float16 v8h;
typedef __attribute__((ext_vector_type(8)))  float    v8f;

static __device__ __forceinline__ float sigmoidf_(float x) {
    return 1.0f / (1.0f + __expf(-x));
}

// ---------------- generic f32 -> f16 cast ----------------
__global__ void cast_f16_kernel(const float* __restrict__ src, _Float16* __restrict__ dst, long n) {
    long i = (long)blockIdx.x * blockDim.x + threadIdx.x;
    if (i < n) dst[i] = (_Float16)src[i];
}

// x_proj_w [DEPTH,56,DI] -> [DEPTH,64,DI] f16, zero pad rows 56..63
__global__ void pad_xp_kernel(const float* __restrict__ src, _Float16* __restrict__ dst) {
    long i = (long)blockIdx.x * blockDim.x + threadIdx.x;
    long n = (long)DEPTH * NXP * DI;
    if (i >= n) return;
    int k = i % DI;
    int r = (i / DI) % NXP;
    int ly = i / ((long)DI * NXP);
    float v = (r < DTR + 2 * DS) ? src[((long)ly * (DTR + 2 * DS) + r) * DI + k] : 0.0f;
    dst[i] = (_Float16)v;
}

// dt_proj_w [DEPTH,DI,24] -> [DEPTH,DI,32] f16, zero pad k 24..31
__global__ void pad_dt_kernel(const float* __restrict__ src, _Float16* __restrict__ dst) {
    long i = (long)blockIdx.x * blockDim.x + threadIdx.x;
    long n = (long)DEPTH * DI * KDT;
    if (i >= n) return;
    int k = i % KDT;
    int r = (i / KDT) % DI;
    int ly = i / ((long)KDT * DI);
    float v = (k < DTR) ? src[((long)ly * DI + r) * DTR + k] : 0.0f;
    dst[i] = (_Float16)v;
}

// im2col of input image -> f16 matrix [BP, KIM]
__global__ void im2col_kernel(const float* __restrict__ x, _Float16* __restrict__ dst) {
    long i = (long)blockIdx.x * blockDim.x + threadIdx.x;
    long n = (long)BP * KIM;
    if (i >= n) return;
    int kk = i % KIM;
    int p  = i / KIM;
    int b  = p / NPATCH, pi = p % NPATCH;
    int gy = pi / GRID,  gx = pi % GRID;
    int c  = kk / (PTCH * PTCH);
    int rem = kk % (PTCH * PTCH);
    int iy = rem / PTCH, ix = rem % PTCH;
    dst[i] = (_Float16)x[(((long)b * C_IN + c) * IMG + gy * PTCH + iy) * IMG + gx * PTCH + ix];
}

// ---------------- fragment loaders (CDNA5 WMMA VGPR layouts) ----------------
// A fragment (16x32, MxK): lane half selects K sub-blocks {0..7,16..23}/{8..15,24..31}
static __device__ __forceinline__ v16h load_afrag(const _Float16* Arow, int k0, int half) {
    v8h lo = *(const v8h*)(Arow + k0 + half * 8);
    v8h hi = *(const v8h*)(Arow + k0 + 16 + half * 8);
    return __builtin_shufflevector(lo, hi,
        0, 1, 2, 3, 4, 5, 6, 7, 8, 9, 10, 11, 12, 13, 14, 15);
}
// B fragment (32x16, KxN): lane half selects K sub-blocks {0..7,16..23}/{8..15,24..31} (col = lane&15)
static __device__ __forceinline__ v16h load_bfrag(const _Float16* Wrow, int k0, int half) {
    v8h lo = *(const v8h*)(Wrow + k0 + half * 8);
    v8h hi = *(const v8h*)(Wrow + k0 + 16 + half * 8);
    return __builtin_shufflevector(lo, hi,
        0, 1, 2, 3, 4, 5, 6, 7, 8, 9, 10, 11, 12, 13, 14, 15);
}

// ---------------- WMMA GEMM: C[M,N] = A[M,K] * W[N,K]^T (+bias)(+resid) ----------------
// A, W: f16 row-major over K.  One wave computes a 32x32 tile (2x2 of 16x16),
// so 4 fragment loads feed 4 v_wmma (16 flops/byte from cache vs 8 for 1x1).
// grid = (N/32, M/32), block = 32 (all lanes active: WMMA needs EXEC all-1s).
__global__ void wmma_gemm_f16_kernel(const _Float16* __restrict__ A,
                                     const _Float16* __restrict__ W,
                                     const float* __restrict__ bias,
                                     const float* __restrict__ resid,
                                     float* __restrict__ C,
                                     int N, int K)
{
    const int ntile = blockIdx.x * 32;
    const int mtile = blockIdx.y * 32;
    const int lane  = threadIdx.x;          // 0..31
    const int half  = lane >> 4;
    const int l15   = lane & 15;

    const _Float16* Arow0 = A + (size_t)(mtile + l15) * K;
    const _Float16* Arow1 = Arow0 + (size_t)16 * K;
    const _Float16* Wrow0 = W + (size_t)(ntile + l15) * K;
    const _Float16* Wrow1 = Wrow0 + (size_t)16 * K;

    v8f acc00 = {}, acc01 = {}, acc10 = {}, acc11 = {};
    for (int k0 = 0; k0 < K; k0 += 32) {
        __builtin_prefetch(Arow0 + k0 + 64, 0, 0);   // global_prefetch_b8
        __builtin_prefetch(Arow1 + k0 + 64, 0, 0);
        v16h a0 = load_afrag(Arow0, k0, half);
        v16h a1 = load_afrag(Arow1, k0, half);
        v16h b0 = load_bfrag(Wrow0, k0, half);
        v16h b1 = load_bfrag(Wrow1, k0, half);
        acc00 = __builtin_amdgcn_wmma_f32_16x16x32_f16(false, a0, false, b0, (short)0, acc00, false, false);
        acc01 = __builtin_amdgcn_wmma_f32_16x16x32_f16(false, a0, false, b1, (short)0, acc01, false, false);
        acc10 = __builtin_amdgcn_wmma_f32_16x16x32_f16(false, a1, false, b0, (short)0, acc10, false, false);
        acc11 = __builtin_amdgcn_wmma_f32_16x16x32_f16(false, a1, false, b1, (short)0, acc11, false, false);
    }

    // C/D layout: VGPR r -> row = r + 8*half, col = lane&15
    const int col0 = ntile + l15;
    const int col1 = col0 + 16;
    const float bv0 = bias ? bias[col0] : 0.0f;
    const float bv1 = bias ? bias[col1] : 0.0f;
#pragma unroll
    for (int r = 0; r < 8; ++r) {
        const int row0 = mtile + half * 8 + r;
        const int row1 = row0 + 16;
        size_t i00 = (size_t)row0 * N + col0;
        size_t i01 = (size_t)row0 * N + col1;
        size_t i10 = (size_t)row1 * N + col0;
        size_t i11 = (size_t)row1 * N + col1;
        float v00 = acc00[r] + bv0;
        float v01 = acc01[r] + bv1;
        float v10 = acc10[r] + bv0;
        float v11 = acc11[r] + bv1;
        if (resid) { v00 += resid[i00]; v01 += resid[i01]; v10 += resid[i10]; v11 += resid[i11]; }
        C[i00] = v00; C[i01] = v01; C[i10] = v10; C[i11] = v11;
    }
}

// ---------------- token assembly: cls + pos ----------------
__global__ void assemble_tok_kernel(const float* __restrict__ feat,   // [BP, D] (patch_b already added)
                                    const float* __restrict__ cls,    // [D]
                                    const float* __restrict__ pos,    // [L, D]
                                    float* __restrict__ tok)          // [BL, D]
{
    long i = (long)blockIdx.x * blockDim.x + threadIdx.x;
    long n = (long)BL * D;
    if (i >= n) return;
    int d = i % D;
    int row = i / D;
    int t = row % L, b = row / L;
    float v;
    if (t == 0) v = cls[d];
    else        v = feat[((long)b * NPATCH + (t - 1)) * D + d];
    tok[i] = v + pos[(long)t * D + d];
}

// ---------------- LayerNorm (one block / row) ----------------
// row_mult lets the final LN pick only cls rows (row = blockIdx.x * 197).
__global__ void ln_kernel(const float* __restrict__ in, int row_mult,
                          const float* __restrict__ w, const float* __restrict__ b,
                          _Float16* __restrict__ out16, float* __restrict__ out32)
{
    constexpr int TPB = 128;
    const int row = blockIdx.x * row_mult;
    const float* r = in + (size_t)row * D;
    __shared__ float red[TPB];
    const int tid = threadIdx.x;

    float s = 0.0f;
    for (int j = tid; j < D; j += TPB) s += r[j];
    red[tid] = s; __syncthreads();
    for (int st = TPB / 2; st > 0; st >>= 1) { if (tid < st) red[tid] += red[tid + st]; __syncthreads(); }
    const float mu = red[0] / (float)D;
    __syncthreads();

    float v = 0.0f;
    for (int j = tid; j < D; j += TPB) { float t = r[j] - mu; v += t * t; }
    red[tid] = v; __syncthreads();
    for (int st = TPB / 2; st > 0; st >>= 1) { if (tid < st) red[tid] += red[tid + st]; __syncthreads(); }
    const float rstd = rsqrtf(red[0] / (float)D + 1e-5f);

    const size_t orow = (size_t)blockIdx.x * D;
    for (int j = tid; j < D; j += TPB) {
        float val = (r[j] - mu) * rstd * w[j] + b[j];
        if (out16) out16[orow + j] = (_Float16)val;
        else       out32[orow + j] = val;
    }
}

// ---------------- causal depthwise conv1d + bias + SiLU ----------------
__global__ void conv_silu_kernel(const float* __restrict__ xz,   // [BL, 2*DI], xb part = [:,0:DI]
                                 const float* __restrict__ cw,   // [DI, DCONV]
                                 const float* __restrict__ cb,   // [DI]
                                 float* __restrict__ xbf,        // [BL, DI]
                                 _Float16* __restrict__ xbh)     // [BL, DI]
{
    long i = (long)blockIdx.x * blockDim.x + threadIdx.x;
    long n = (long)BL * DI;
    if (i >= n) return;
    int d = i % DI;
    long bl = i / DI;
    int l = bl % L, b = bl / L;
    float acc = cb[d];
#pragma unroll
    for (int j = 0; j < DCONV; ++j) {
        int ls = l - (DCONV - 1) + j;
        if (ls >= 0)
            acc += cw[d * DCONV + j] * xz[((size_t)b * L + ls) * (2 * DI) + d];
    }
    float sv = acc * sigmoidf_(acc);
    xbf[i] = sv;
    xbh[i] = (_Float16)sv;
}

// build dt-GEMM input [BL, 32] f16 from xdbl[:, 0:24], zero padded
__global__ void build_dt_in_kernel(const float* __restrict__ xdbl, _Float16* __restrict__ dtin) {
    long i = (long)blockIdx.x * blockDim.x + threadIdx.x;
    long n = (long)BL * KDT;
    if (i >= n) return;
    int k = i % KDT;
    long row = i / KDT;
    float v = (k < DTR) ? xdbl[row * NXP + k] : 0.0f;
    dtin[i] = (_Float16)v;
}

// ---------------- selective scan (one block per batch, thread = channel d) ----------------
__global__ void scan_kernel(const float* __restrict__ xz,     // [BL, 2*DI]  (z part)
                            const float* __restrict__ xbf,    // [BL, DI]
                            const float* __restrict__ xdbl,   // [BL, 64]    B at 24..39, C at 40..55
                            const float* __restrict__ dtlin,  // [BL, DI]
                            const float* __restrict__ dtb,    // [DI]
                            const float* __restrict__ A_log,  // [DI, DS]
                            const float* __restrict__ Dp,     // [DI]
                            _Float16* __restrict__ yout)      // [BL, DI]
{
    const int b = blockIdx.x;
    const int d = threadIdx.x;     // 0..DI-1 (768 threads)
    __shared__ float sBC[2 * DS];  // B then C

    float a[DS], h[DS];
#pragma unroll
    for (int s = 0; s < DS; ++s) {
        a[s] = -__expf(A_log[(size_t)d * DS + s]);
        h[s] = 0.0f;
    }
    const float dtbv = dtb[d];
    const float Dpd  = Dp[d];

    for (int l = 0; l < L; ++l) {
        const size_t row = (size_t)b * L + l;
        if (d < 2 * DS) sBC[d] = xdbl[row * NXP + DTR + d];
        __syncthreads();

        float dt = dtlin[row * DI + d] + dtbv;
        dt = (dt > 20.0f) ? dt : log1pf(__expf(dt));     // softplus
        const float xv = xbf[row * DI + d];
        float y = 0.0f;
#pragma unroll
        for (int s = 0; s < DS; ++s) {
            const float da = __expf(dt * a[s]);
            h[s] = da * h[s] + dt * sBC[s] * xv;
            y += h[s] * sBC[DS + s];
        }
        y += Dpd * xv;
        const float z = xz[row * (2 * DI) + DI + d];
        yout[row * DI + d] = (_Float16)(y * (z * sigmoidf_(z)));
        __syncthreads();
    }
}

// ---------------- classifier head ----------------
__global__ void head_kernel(const float* __restrict__ lncls,   // [BATCH, D]
                            const float* __restrict__ hw,      // [NCLS, D]
                            const float* __restrict__ hb,      // [NCLS]
                            float* __restrict__ out)           // [BATCH, NCLS]
{
    int i = blockIdx.x * blockDim.x + threadIdx.x;
    if (i >= BATCH * NCLS) return;
    int n = i % NCLS, b = i / NCLS;
    float acc = hb[n];
    for (int k = 0; k < D; ++k) acc += lncls[(size_t)b * D + k] * hw[(size_t)n * D + k];
    out[i] = acc;
}

// ---------------- host orchestration ----------------
static inline long cdiv(long a, long b) { return (a + b - 1) / b; }

extern "C" void kernel_launch(void* const* d_in, const int* in_sizes, int n_in,
                              void* d_out, int out_size, void* d_ws, size_t ws_size,
                              hipStream_t stream) {
    (void)in_sizes; (void)n_in; (void)out_size; (void)ws_size;
    const float* x        = (const float*)d_in[0];
    const float* patch_w  = (const float*)d_in[1];
    const float* patch_b  = (const float*)d_in[2];
    const float* cls_tok  = (const float*)d_in[3];
    const float* pos_emb  = (const float*)d_in[4];
    const float* ln_w     = (const float*)d_in[5];
    const float* ln_b     = (const float*)d_in[6];
    const float* in_w     = (const float*)d_in[7];
    const float* conv_w   = (const float*)d_in[8];
    const float* conv_b   = (const float*)d_in[9];
    const float* xp_w     = (const float*)d_in[10];
    const float* dt_w     = (const float*)d_in[11];
    const float* dt_b     = (const float*)d_in[12];
    const float* A_log    = (const float*)d_in[13];
    const float* D_param  = (const float*)d_in[14];
    const float* out_w    = (const float*)d_in[15];
    const float* normf_w  = (const float*)d_in[16];
    const float* normf_b  = (const float*)d_in[17];
    const float* head_w   = (const float*)d_in[18];
    const float* head_b   = (const float*)d_in[19];
    float* out = (float*)d_out;

    // -------- workspace carve-up --------
    char* base = (char*)d_ws;
    size_t off = 0;
    auto carve = [&](size_t bytes) -> char* {
        char* p = base + off;
        off += (bytes + 255) & ~(size_t)255;
        return p;
    };
    float*    tok    = (float*)   carve((size_t)BL * D * 4);
    _Float16* xln    = (_Float16*)carve((size_t)BL * D * 2);
    float*    xz     = (float*)   carve((size_t)BL * 2 * DI * 4);   // also holds feat [BP,D] early
    float*    xbf    = (float*)   carve((size_t)BL * DI * 4);
    _Float16* xbh    = (_Float16*)carve((size_t)BL * DI * 2);       // also holds im2col early
    float*    xdbl   = (float*)   carve((size_t)BL * NXP * 4);
    _Float16* dtin   = (_Float16*)carve((size_t)BL * KDT * 2);
    float*    dtlin  = (float*)   carve((size_t)BL * DI * 4);
    _Float16* yh     = (_Float16*)carve((size_t)BL * DI * 2);
    float*    lncls  = (float*)   carve((size_t)BATCH * D * 4);
    _Float16* w_in   = (_Float16*)carve((size_t)DEPTH * 2 * DI * D * 2);
    _Float16* w_out  = (_Float16*)carve((size_t)DEPTH * D * DI * 2);
    _Float16* w_xp   = (_Float16*)carve((size_t)DEPTH * NXP * DI * 2);
    _Float16* w_dt   = (_Float16*)carve((size_t)DEPTH * DI * KDT * 2);
    _Float16* w_pat  = (_Float16*)carve((size_t)D * KIM * 2);

    float*    feat   = xz;                    // [BP, D] reuse
    _Float16* imcol  = xbh;                   // [BP, KIM] reuse

    const int TPB = 256;

    // -------- one-time (per launch) weight casts --------
    {
        long n = (long)D * KIM;
        cast_f16_kernel<<<cdiv(n, TPB), TPB, 0, stream>>>(patch_w, w_pat, n);
        n = (long)DEPTH * 2 * DI * D;
        cast_f16_kernel<<<cdiv(n, TPB), TPB, 0, stream>>>(in_w, w_in, n);
        n = (long)DEPTH * D * DI;
        cast_f16_kernel<<<cdiv(n, TPB), TPB, 0, stream>>>(out_w, w_out, n);
        n = (long)DEPTH * NXP * DI;
        pad_xp_kernel<<<cdiv(n, TPB), TPB, 0, stream>>>(xp_w, w_xp);
        n = (long)DEPTH * DI * KDT;
        pad_dt_kernel<<<cdiv(n, TPB), TPB, 0, stream>>>(dt_w, w_dt);
    }

    // -------- patch embedding (im2col + WMMA GEMM) --------
    {
        long n = (long)BP * KIM;
        im2col_kernel<<<cdiv(n, TPB), TPB, 0, stream>>>(x, imcol);
        dim3 grid(D / 32, BP / 32);
        wmma_gemm_f16_kernel<<<grid, 32, 0, stream>>>(imcol, w_pat, patch_b, nullptr,
                                                      feat, D, KIM);
        n = (long)BL * D;
        assemble_tok_kernel<<<cdiv(n, TPB), TPB, 0, stream>>>(feat, cls_tok, pos_emb, tok);
    }

    // -------- mamba layers --------
    for (int ly = 0; ly < DEPTH; ++ly) {
        // LN -> f16
        ln_kernel<<<BL, 128, 0, stream>>>(tok, 1, ln_w + (size_t)ly * D,
                                          ln_b + (size_t)ly * D, xln, nullptr);
        // in_proj: [BL,D] x [2DI,D]^T -> xz [BL,2DI]
        {
            dim3 grid((2 * DI) / 32, BL / 32);
            wmma_gemm_f16_kernel<<<grid, 32, 0, stream>>>(
                xln, w_in + (size_t)ly * 2 * DI * D, nullptr, nullptr, xz, 2 * DI, D);
        }
        // depthwise causal conv + SiLU
        {
            long n = (long)BL * DI;
            conv_silu_kernel<<<cdiv(n, TPB), TPB, 0, stream>>>(
                xz, conv_w + (size_t)ly * DI * DCONV, conv_b + (size_t)ly * DI, xbf, xbh);
        }
        // x_proj: [BL,DI] x [64,DI]^T -> xdbl [BL,64]
        {
            dim3 grid(NXP / 32, BL / 32);
            wmma_gemm_f16_kernel<<<grid, 32, 0, stream>>>(
                xbh, w_xp + (size_t)ly * NXP * DI, nullptr, nullptr, xdbl, NXP, DI);
        }
        // dt_proj: pad K, then [BL,32] x [DI,32]^T -> dtlin [BL,DI]
        {
            long n = (long)BL * KDT;
            build_dt_in_kernel<<<cdiv(n, TPB), TPB, 0, stream>>>(xdbl, dtin);
            dim3 grid(DI / 32, BL / 32);
            wmma_gemm_f16_kernel<<<grid, 32, 0, stream>>>(
                dtin, w_dt + (size_t)ly * DI * KDT, nullptr, nullptr, dtlin, DI, KDT);
        }
        // selective scan (fused softplus, dA, state update, C-dot, D-skip, SiLU gate)
        scan_kernel<<<BATCH, DI, 0, stream>>>(
            xz, xbf, xdbl, dtlin,
            dt_b + (size_t)ly * DI,
            A_log + (size_t)ly * DI * DS,
            D_param + (size_t)ly * DI, yh);
        // out_proj + residual (in place on tok)
        {
            dim3 grid(D / 32, BL / 32);
            wmma_gemm_f16_kernel<<<grid, 32, 0, stream>>>(
                yh, w_out + (size_t)ly * D * DI, nullptr, tok, tok, D, DI);
        }
    }

    // -------- final LN (cls rows only) + head --------
    ln_kernel<<<BATCH, 128, 0, stream>>>(tok, L, normf_w, normf_b, nullptr, lncls);
    head_kernel<<<cdiv((long)BATCH * NCLS, TPB), TPB, 0, stream>>>(lncls, head_w, head_b, out);
}
